// MixSoftmax_77790447665589
// MI455X (gfx1250) — matrix operand order repaired and checked by reading, not compiled, not run on hardware
//
#include <hip/hip_runtime.h>
#include <math.h>

#define MARGIN 0.35f
#define T_HARD 0.2f
#define SCALE  32.0f
#define EPS    1e-7f
#define CMAX   44.8f   // max possible logit: 32*((1+t)*1 + t)

#define EMBED 512
#define BATCH 512
#define NBLK  64                 // classes per GEMM block
#define LRSTR (4 * 2 * EMBED + 8) // 4104 f16 per lr slot: [4 ct][2 part][512] + 8 pad
                                  // -> lane stride 8208 B = 2052 dwords === 4 (mod 64)
#define KCH   32                 // TDM K-chunk (f32 columns per chunk)
#define NCH   (EMBED / KCH)      // 16 chunks

typedef _Float16 h2  __attribute__((ext_vector_type(2)));
typedef _Float16 h8  __attribute__((ext_vector_type(8)));
typedef _Float16 h16 __attribute__((ext_vector_type(16)));
typedef float    v8f __attribute__((ext_vector_type(8)));
typedef unsigned int u32x4 __attribute__((ext_vector_type(4)));
typedef int          i32x4 __attribute__((ext_vector_type(4)));
typedef int          i32x8 __attribute__((ext_vector_type(8)));

union H16 { h16 v; h8 h[2]; };

#if defined(__AMDGCN__) && __has_builtin(__builtin_amdgcn_tensor_load_to_lds)
#define USE_TDM 1
#else
#define USE_TDM 0
#endif

// W tile index (f16 units), lr-major: everything lane lr reads lives in one
// 8192-byte window -> all hot-loop DS offsets are immediates off one base.
__host__ __device__ __forceinline__ int widx(int cls, int part, int k) {
  return (cls & 15) * LRSTR + (cls >> 4) * (2 * EMBED) + part * EMBED + k;
}

// LDS layout (bytes) — identical for both phase-1 paths
#define LDS_WT    0                                    // 16*LRSTR f16 = 131328 B
#define LDS_STG   (16 * LRSTR * 2)                     // 131328 (2 x 8KB f32 staging)
#define LDS_INW   (LDS_STG + 2 * NBLK * KCH * 4)       // 147712
#define LDS_PSUM  (LDS_INW + NBLK * 4)                 // 147968
#define LDS_GTC   (LDS_PSUM + NBLK * 4 * 4)            // 148992
#define LDS_TOTAL (LDS_GTC + BATCH * 4)                // 151040

// ---------------------------------------------------------------------------
// Kernel 1: per-row embedding norm, split-f16 (hi/lo) normalized embedding,
//           exact f32 clipped target cosine per row.
// ---------------------------------------------------------------------------
__global__ void prep_kernel(const float* __restrict__ emb,
                            const int*   __restrict__ gt,
                            const float* __restrict__ wgt,
                            _Float16* __restrict__ e_hi,
                            _Float16* __restrict__ e_lo,
                            float* __restrict__ gt_cos) {
  const int r = blockIdx.x;
  const int t = threadIdx.x;                 // 256 threads
  const float e0 = emb[r * EMBED + t];
  const float e1 = emb[r * EMBED + 256 + t];
  const int   g  = gt[r];
  const float w0 = wgt[(size_t)g * EMBED + t];
  const float w1 = wgt[(size_t)g * EMBED + 256 + t];

  float ss = e0 * e0 + e1 * e1;
  float ws = w0 * w0 + w1 * w1;
  float dt = e0 * w0 + e1 * w1;
  for (int o = 16; o > 0; o >>= 1) {
    ss += __shfl_down(ss, o);
    ws += __shfl_down(ws, o);
    dt += __shfl_down(dt, o);
  }
  __shared__ float red[8][3];
  __shared__ float tot[3];
  const int wv = t >> 5, ln = t & 31;
  if (ln == 0) { red[wv][0] = ss; red[wv][1] = ws; red[wv][2] = dt; }
  __syncthreads();
  if (t == 0) {
    float a = 0.f, b = 0.f, c = 0.f;
    for (int i = 0; i < 8; ++i) { a += red[i][0]; b += red[i][1]; c += red[i][2]; }
    tot[0] = a; tot[1] = b; tot[2] = c;
  }
  __syncthreads();
  const float inv_e = 1.0f / sqrtf(tot[0]);
  if (t == 0) {
    float cg = tot[2] * inv_e * (1.0f / sqrtf(tot[1]));
    cg = fminf(fmaxf(cg, -1.0f + EPS), 1.0f - EPS);
    gt_cos[r] = cg;
  }
  const float x0 = e0 * inv_e, x1 = e1 * inv_e;
  const _Float16 h0 = (_Float16)x0, h1 = (_Float16)x1;
  e_hi[r * EMBED + t]       = h0;
  e_hi[r * EMBED + 256 + t] = h1;
  e_lo[r * EMBED + t]       = (_Float16)(x0 - (float)h0);
  e_lo[r * EMBED + 256 + t] = (_Float16)(x1 - (float)h1);
}

// ---------------------------------------------------------------------------
// TDM helper: issue a Tensor-DMA load of one W K-chunk (64 rows x KCH f32)
// into LDS staging. Built per ISA 8.3/8.4 D# bitfields. Wave-0 only.
// ---------------------------------------------------------------------------
__device__ __forceinline__ void tdm_issue(const float* gsrc, unsigned lds_off,
                                          unsigned rows_remaining) {
#if USE_TDM
  const unsigned long long ga = (unsigned long long)(size_t)gsrc;
  u32x4 g0;
  g0.x = 1u;                                  // count=1 (valid), user mode
  g0.y = lds_off;                             // lds_addr (bytes)
  g0.z = (unsigned)(ga & 0xffffffffu);        // global_addr[31:0]
  g0.w = (unsigned)((ga >> 32) & 0x1ffffffu)  // global_addr[56:32]
         | (2u << 30);                        // type=2 ("image")
  i32x8 g1;
  g1.s0 = (int)(2u << 16);                    // data_size=2 -> 4 bytes
  g1.s1 = (int)((unsigned)EMBED << 16);       // tensor_dim0[15:0] (=512)
  g1.s2 = (int)((rows_remaining & 0xffffu) << 16); // dim0[31:16]=0 | tensor_dim1 lo
  g1.s3 = (int)((rows_remaining >> 16) | ((unsigned)KCH << 16)); // dim1 hi | tile_dim0
  g1.s4 = (int)NBLK;                          // tile_dim1=64 | tile_dim2=0
  g1.s5 = (int)EMBED;                         // tensor_dim0_stride[31:0]
  g1.s6 = 0;                                  // stride0 hi | stride1 lo
  g1.s7 = 0;
  i32x4 gz = {0, 0, 0, 0};
#if __clang_major__ >= 23
  i32x8 gz8 = {0, 0, 0, 0, 0, 0, 0, 0};
  __builtin_amdgcn_tensor_load_to_lds(g0, g1, gz, gz, gz8, 0);
#else
  __builtin_amdgcn_tensor_load_to_lds(g0, g1, gz, gz, 0);
#endif
#else
  (void)gsrc; (void)lds_off; (void)rows_remaining;
#endif
}

// ---------------------------------------------------------------------------
// Kernel 2: fused cosine-GEMM (split-f16 WMMA) + MV-AM transform + fixed-max
//           exp-sum. Block = 64 classes x 512 rows x K=512.
//           Wave w: rows [w*64, w*64+64), ALL 64 classes (2 Mtiles x 4 Ntiles).
// ---------------------------------------------------------------------------
__global__ void gemm_softmax_kernel(const float* __restrict__ wgt,
                                    const _Float16* __restrict__ e_hi,
                                    const _Float16* __restrict__ e_lo,
                                    const float* __restrict__ gt_cos,
                                    float* __restrict__ partials,
                                    int C, int NB) {
  extern __shared__ char smem[];
  _Float16* wt    = (_Float16*)(smem + LDS_WT);
  float*    stg   = (float*)(smem + LDS_STG);          // 2 x (64 x KCH) f32
  float*    invnw = (float*)(smem + LDS_INW);
  float*    psum  = (float*)(smem + LDS_PSUM);
  float*    gtc   = (float*)(smem + LDS_GTC);

  const int t  = threadIdx.x;                          // 256 threads
  const int b  = blockIdx.x;
  const int c0 = b * NBLK;
  const int wv = t >> 5;
  const int lane = t & 31;

  gtc[t]       = gt_cos[t];
  gtc[t + 256] = gt_cos[t + 256];

#if USE_TDM
  // -------- Phase 1a (TDM): double-buffered async tensor loads ------------
  const unsigned rows_rem = (unsigned)(C - c0);        // OOB rows read as 0
  const float* wbase = wgt + (size_t)c0 * EMBED;
  if (wv == 0) tdm_issue(wbase, (unsigned)(size_t)stg, rows_rem);
  for (int c = 0; c < NCH; ++c) {
    if (wv == 0) {
      if (c + 1 < NCH) {
        tdm_issue(wbase + (c + 1) * KCH,
                  (unsigned)(size_t)(stg + ((c + 1) & 1) * (NBLK * KCH)),
                  rows_rem);
        __builtin_amdgcn_s_wait_tensorcnt(1);          // chunk c complete
      } else {
        __builtin_amdgcn_s_wait_tensorcnt(0);
      }
    }
    __syncthreads();                                   // publish chunk c
    {
      const float* s = stg + (c & 1) * (NBLK * KCH);
      const int r = t >> 2, q = t & 3;                 // 4 threads per class row
      const int col = c * KCH + q * 8;
      _Float16* dh = wt + widx(r, 0, col);
      _Float16* dl = wt + widx(r, 1, col);
#pragma unroll
      for (int u = 0; u < 2; ++u) {
        const float4 x = *(const float4*)(s + r * KCH + q * 8 + 4 * u);
        const _Float16 a0 = (_Float16)x.x, a1 = (_Float16)x.y;
        const _Float16 a2 = (_Float16)x.z, a3 = (_Float16)x.w;
        h2 ph0 = {a0, a1}, ph1 = {a2, a3};
        *(h2*)(dh + 4 * u)     = ph0;
        *(h2*)(dh + 4 * u + 2) = ph1;
        h2 pl0 = {(_Float16)(x.x - (float)a0), (_Float16)(x.y - (float)a1)};
        h2 pl1 = {(_Float16)(x.z - (float)a2), (_Float16)(x.w - (float)a3)};
        *(h2*)(dl + 4 * u)     = pl0;
        *(h2*)(dl + 4 * u + 2) = pl1;
      }
    }
    __syncthreads();                                   // staging reusable
  }
#else
  // -------- Phase 1a (fallback): direct global loads ----------------------
  for (int idx = t; idx < NBLK * (EMBED / 4); idx += 256) {
    const int r = idx >> 7;
    const int f = idx & 127;
    const int g = c0 + r;
    float4 x = make_float4(0.f, 0.f, 0.f, 0.f);
    if (g < C) {
      const float* p = wgt + (size_t)g * EMBED + 4 * f;
      __builtin_prefetch(p + 256, 0, 1);
      x = *(const float4*)p;
    }
    const _Float16 a0 = (_Float16)x.x, a1 = (_Float16)x.y;
    const _Float16 a2 = (_Float16)x.z, a3 = (_Float16)x.w;
    h2 ph0 = {a0, a1}, ph1 = {a2, a3};
    _Float16* dh = wt + widx(r, 0, 4 * f);
    _Float16* dl = wt + widx(r, 1, 4 * f);
    *(h2*)(dh)     = ph0;
    *(h2*)(dh + 2) = ph1;
    h2 pl0 = {(_Float16)(x.x - (float)a0), (_Float16)(x.y - (float)a1)};
    h2 pl1 = {(_Float16)(x.z - (float)a2), (_Float16)(x.w - (float)a3)};
    *(h2*)(dl)     = pl0;
    *(h2*)(dl + 2) = pl1;
  }
  __syncthreads();
#endif

  // -------- Phase 1b: deterministic per-class inverse norms ---------------
  {
    const int r = t >> 2, q = t & 3;
    const int bh = widx(r, 0, 0), bl = widx(r, 1, 0);
    float s = 0.f;
    for (int k = q * 128; k < q * 128 + 128; ++k) {
      const float x = (float)wt[bh + k] + (float)wt[bl + k];
      s += x * x;
    }
    psum[r * 4 + q] = s;
  }
  __syncthreads();
  if (t < NBLK) {
    const float s = psum[t * 4] + psum[t * 4 + 1] + psum[t * 4 + 2] + psum[t * 4 + 3];
    invnw[t] = (s > 0.f) ? (1.0f / sqrtf(s)) : 0.f;
  }
  __syncthreads();

  // -------- Phase 2: register-blocked WMMA GEMM + post-process ------------
  const int lr   = lane & 15;
  const int hi16 = lane >> 4;                          // 0 or 1
  float inw[4];
  bool  val[4];
#pragma unroll
  for (int ct = 0; ct < 4; ++ct) {
    inw[ct] = invnw[ct * 16 + lr];
    val[ct] = (c0 + ct * 16 + lr) < C;
  }
  // B operand (32x16 KxN): lane n<16 -> col n, K {0-7,16-23}; lane>=16 -> K {8-15,24-31}.
  // Single base pointer per lane; ALL loop offsets are DS immediates <= 8176B:
  //   + ct*2*EMBED + part*EMBED + k0 (+16)
  const _Float16* bb = wt + lr * LRSTR + hi16 * 8;

  const int rbase = wv * 64;                           // 64 rows per wave
#pragma unroll 1
  for (int half = 0; half < 2; ++half) {
    const int m0 = rbase + half * 32;
    // A operand (16x32): lane<16 -> K {0-7,16-23}; lane>=16 -> K {8-15,24-31}
    const _Float16* a_hi = e_hi + (m0 + lr) * EMBED + hi16 * 8;
    const _Float16* a_lo = e_lo + (m0 + lr) * EMBED + hi16 * 8;

    v8f acc[2][4];
#pragma unroll
    for (int m = 0; m < 2; ++m)
#pragma unroll
      for (int ct = 0; ct < 4; ++ct) acc[m][ct] = (v8f){};

#pragma unroll 4
    for (int ks = 0; ks < 16; ++ks) {
      const int k0 = ks * 32;
      H16 Ah[2], Al[2];
#pragma unroll
      for (int m = 0; m < 2; ++m) {
        const _Float16* ph = a_hi + m * 16 * EMBED + k0;
        const _Float16* pl = a_lo + m * 16 * EMBED + k0;
        Ah[m].h[0] = *(const h8*)(ph);
        Ah[m].h[1] = *(const h8*)(ph + 16);
        Al[m].h[0] = *(const h8*)(pl);
        Al[m].h[1] = *(const h8*)(pl + 16);
      }
#pragma unroll
      for (int ct = 0; ct < 4; ++ct) {
        const _Float16* bp = bb + ct * (2 * EMBED);
        H16 Bh, Bl;
        Bh.h[0] = *(const h8*)(bp + k0);
        Bh.h[1] = *(const h8*)(bp + k0 + 16);
        Bl.h[0] = *(const h8*)(bp + EMBED + k0);
        Bl.h[1] = *(const h8*)(bp + EMBED + k0 + 16);
#pragma unroll
        for (int m = 0; m < 2; ++m) {
          // split-f32 dot: hi*hi + hi*lo + lo*hi (error ~2^-22)
          acc[m][ct] = __builtin_amdgcn_wmma_f32_16x16x32_f16(
              false, Ah[m].v, false, Bh.v, (short)0, acc[m][ct], false, false);
          acc[m][ct] = __builtin_amdgcn_wmma_f32_16x16x32_f16(
              false, Ah[m].v, false, Bl.v, (short)0, acc[m][ct], false, false);
          acc[m][ct] = __builtin_amdgcn_wmma_f32_16x16x32_f16(
              false, Al[m].v, false, Bh.v, (short)0, acc[m][ct], false, false);
        }
      }
    }

    // Post-process: full per-row sum over this block's 64 classes.
#pragma unroll
    for (int m = 0; m < 2; ++m) {
#pragma unroll
      for (int i = 0; i < 8; ++i) {
        const int row = m0 + m * 16 + hi16 * 8 + i;    // C/D: lane<16 M=i else M=8+i
        const float gv = gtc[row];
        float ex = 0.f;
#pragma unroll
        for (int ct = 0; ct < 4; ++ct) {
          float d = acc[m][ct][i] * inw[ct];
          d = fminf(fmaxf(d, -1.0f + EPS), 1.0f - EPS);
          if (d > gv - MARGIN) d = (1.0f + T_HARD) * d + T_HARD;
          ex += val[ct] ? __expf(SCALE * d - CMAX) : 0.0f;
        }
        ex += __shfl_xor(ex, 1);
        ex += __shfl_xor(ex, 2);
        ex += __shfl_xor(ex, 4);
        ex += __shfl_xor(ex, 8);
        if (lr == 0) partials[(size_t)row * NB + b] = ex;
      }
    }
  }
}

// ---------------------------------------------------------------------------
// Kernel 3: deterministic reduction + target-column correction -> row loss.
// ---------------------------------------------------------------------------
__global__ void rowloss_kernel(const float* __restrict__ partials,
                               const float* __restrict__ gt_cos,
                               float* __restrict__ rowloss, int NB) {
  const int r = blockIdx.x;
  const int t = threadIdx.x;
  float s = 0.f;
  for (int j = t; j < NB; j += 256) s += partials[(size_t)r * NB + j];
  __shared__ float red[256];
  red[t] = s;
  __syncthreads();
  for (int o = 128; o > 0; o >>= 1) {
    if (t < o) red[t] += red[t + o];
    __syncthreads();
  }
  if (t == 0) {
    const float g = gt_cos[r];
    float total = red[0];
    total -= expf(SCALE * ((1.0f + T_HARD) * g + T_HARD) - CMAX);
    total += expf(SCALE * (g - MARGIN) - CMAX);
    const float lse = CMAX + logf(total);
    rowloss[r] = lse - SCALE * (g - MARGIN);
  }
}

__global__ void mean_kernel(const float* __restrict__ rowloss,
                            float* __restrict__ out) {
  const int t = threadIdx.x;
  float s = rowloss[t] + rowloss[t + 256];
  __shared__ float red[256];
  red[t] = s;
  __syncthreads();
  for (int o = 128; o > 0; o >>= 1) {
    if (t < o) red[t] += red[t + o];
    __syncthreads();
  }
  if (t == 0) out[0] = red[0] / (float)BATCH;
}

// ---------------------------------------------------------------------------
extern "C" void kernel_launch(void* const* d_in, const int* in_sizes, int n_in,
                              void* d_out, int out_size, void* d_ws, size_t ws_size,
                              hipStream_t stream) {
  const float* emb = (const float*)d_in[0];
  const int*   gt  = (const int*)d_in[1];
  const float* wgt = (const float*)d_in[2];
  const int C  = in_sizes[2] / EMBED;            // 72690
  const int NB = (C + NBLK - 1) / NBLK;          // 1136 class-blocks

  char* ws = (char*)d_ws;
  _Float16* e_hi   = (_Float16*)ws;              // BATCH*EMBED f16
  _Float16* e_lo   = e_hi + BATCH * EMBED;
  float*    gt_cos = (float*)(e_lo + BATCH * EMBED);
  float*    rloss  = gt_cos + BATCH;
  float*    parts  = rloss + BATCH;              // BATCH*NB f32

  prep_kernel<<<BATCH, 256, 0, stream>>>(emb, gt, wgt, e_hi, e_lo, gt_cos);

  (void)hipFuncSetAttribute((const void*)gemm_softmax_kernel,
                            hipFuncAttributeMaxDynamicSharedMemorySize,
                            (int)LDS_TOTAL);
  gemm_softmax_kernel<<<NB, 256, LDS_TOTAL, stream>>>(wgt, e_hi, e_lo, gt_cos,
                                                      parts, C, NB);

  rowloss_kernel<<<BATCH, 256, 0, stream>>>(parts, gt_cos, rloss, NB);
  mean_kernel<<<1, 256, 0, stream>>>(rloss, (float*)d_out);
}
